// FinalLayer_14474039787746
// MI455X (gfx1250) — 8-bit matrix operand order repaired and checked by reading, compiled, not run on hardware
//
#include <hip/hip_runtime.h>
#include <math.h>


typedef int   v8i __attribute__((ext_vector_type(8)));
typedef int   v4i __attribute__((ext_vector_type(4)));
typedef float v4f __attribute__((ext_vector_type(4)));

#define HDIM   1024
#define OUTN   256
#define BATCH  8
#define SEQ    4096
#define EPSLN  1e-6f
#define MTILE  16
#define LDS_PITCH 1040

__global__ void __launch_bounds__(256) k_wabs_sum(const float* __restrict__ w,
                                                  float* __restrict__ sum_ws) {
    __shared__ float red[256];
    float s = 0.f;
    for (int i = threadIdx.x; i < OUTN * HDIM; i += 256) s += fabsf(w[i]);
    red[threadIdx.x] = s;
    __syncthreads();
    for (int off = 128; off > 0; off >>= 1) {
        if ((int)threadIdx.x < off) red[threadIdx.x] += red[threadIdx.x + off];
        __syncthreads();
    }
    if (threadIdx.x == 0) sum_ws[0] = red[0];
}

__global__ void __launch_bounds__(256) k_wquant_swizzle(const float* __restrict__ w,
                                                        const float* __restrict__ sum_ws,
                                                        unsigned* __restrict__ wq) {
    const int tid  = blockIdx.x * 256 + threadIdx.x;
    const int d    = tid & 255;
    const int f    = tid >> 8;
    const int lane = (d >> 2) & 31;
    const int v    = ((d >> 7) << 2) | (d & 3);
    const int kstep = f & 15;
    const int ntile = f >> 4;
    const int n     = ntile * 16 + (lane & 15);
    const int half  = lane >> 4;
    const int kbase = kstep * 64 + (v & 3) * 4 + half * 16 + (v >> 2) * 32;

    const float mean  = fmaxf(sum_ws[0] * (1.0f / (OUTN * HDIM)), 1e-5f);
    const float scale = 1.0f / mean;

    unsigned pack = 0u;
#pragma unroll
    for (int j = 0; j < 4; ++j) {
        float q = rintf(w[n * HDIM + kbase + j] * scale);
        q = fminf(fmaxf(q, -1.f), 1.f);
        pack |= ((unsigned)(((int)q) & 0xFF)) << (8 * j);
    }
    wq[tid] = pack;
}

__global__ void __launch_bounds__(256) k_adaln_mod(const float* __restrict__ c,
                                                   const float* __restrict__ w_ada,
                                                   const float* __restrict__ b_ada,
                                                   float* __restrict__ modws) {
    const int wave = threadIdx.x >> 5;
    const int lane = threadIdx.x & 31;
    const int b    = blockIdx.x >> 8;
    const int o    = (blockIdx.x & 255) * 8 + wave;
    const float* crow = c + b * HDIM;
    const float* wrow = w_ada + o * HDIM;
    float s = 0.f;
    for (int h = lane; h < HDIM; h += 32) {
        float a  = crow[h];
        float sl = a / (1.0f + expf(-a));
        s += sl * wrow[h];
    }
#pragma unroll
    for (int off = 16; off; off >>= 1) s += __shfl_xor(s, off, 32);
    if (lane == 0) modws[b * 2048 + o] = s + b_ada[o];
}

__global__ void __launch_bounds__(256) k_final_bitlinear(const float* __restrict__ x,
                                                         const float* __restrict__ modws,
                                                         const unsigned* __restrict__ wq,
                                                         const float* __restrict__ sum_ws,
                                                         const float* __restrict__ b_lin,
                                                         float* __restrict__ out) {
    __shared__ char  qx[MTILE * LDS_PITCH];
    __shared__ float inv_sx[MTILE];

    const int tid     = threadIdx.x;
    const int lane    = tid & 31;
    const int wave    = tid >> 5;
    const int tokbase = blockIdx.x * MTILE;
    const int b       = tokbase / SEQ;
    const float* shiftrow = modws + b * 2048;
    const float* scalerow = shiftrow + HDIM;

    for (int tt = 0; tt < 2; ++tt) {
        const int tok = wave * 2 + tt;
        const float* xrow = x + (size_t)(tokbase + tok) * HDIM;
        v4f xv[8];
        float sum = 0.f, sumsq = 0.f;
#pragma unroll
        for (int i = 0; i < 8; ++i) {
            xv[i] = ((const v4f*)xrow)[i * 32 + lane];
#pragma unroll
            for (int j = 0; j < 4; ++j) { float t = xv[i][j]; sum += t; sumsq += t * t; }
        }
#pragma unroll
        for (int off = 16; off; off >>= 1) {
            sum   += __shfl_xor(sum,   off, 32);
            sumsq += __shfl_xor(sumsq, off, 32);
        }
        const float mu   = sum * (1.0f / HDIM);
        const float var  = sumsq * (1.0f / HDIM) - mu * mu;
        const float rstd = rsqrtf(var + EPSLN);

        float amax = 0.f;
#pragma unroll
        for (int i = 0; i < 8; ++i) {
            v4f sc = ((const v4f*)scalerow)[i * 32 + lane];
            v4f sh = ((const v4f*)shiftrow)[i * 32 + lane];
#pragma unroll
            for (int j = 0; j < 4; ++j) {
                float xm = (xv[i][j] - mu) * rstd * (1.0f + sc[j]) + sh[j];
                xv[i][j] = xm;
                amax = fmaxf(amax, fabsf(xm));
            }
        }
#pragma unroll
        for (int off = 16; off; off >>= 1) amax = fmaxf(amax, __shfl_xor(amax, off, 32));
        amax = fmaxf(amax, 1e-5f);
        const float sx = 127.0f / amax;
        if (lane == 0) inv_sx[tok] = amax * (1.0f / 127.0f);

#pragma unroll
        for (int i = 0; i < 8; ++i) {
            unsigned pack = 0u;
#pragma unroll
            for (int j = 0; j < 4; ++j) {
                float q = fminf(fmaxf(rintf(xv[i][j] * sx), -128.f), 127.f);
                pack |= ((unsigned)(((int)q) & 0xFF)) << (8 * j);
            }
            *(unsigned*)(&qx[tok * LDS_PITCH + i * 128 + lane * 4]) = pack;
        }
    }
    __syncthreads();

    const float wmean = fmaxf(sum_ws[0] * (1.0f / (OUTN * HDIM)), 1e-5f);

    const int m    = lane & 15;
    const int half = lane >> 4;
    v8i acc0 = {0, 0, 0, 0, 0, 0, 0, 0};
    v8i acc1 = {0, 0, 0, 0, 0, 0, 0, 0};

    const char* fragbase0 = (const char*)wq + ((size_t)(wave * 2 + 0) * 16) * 1024 + lane * 16;
    const char* fragbase1 = (const char*)wq + ((size_t)(wave * 2 + 1) * 16) * 1024 + lane * 16;

    for (int kstep = 0; kstep < 16; ++kstep) {
        v8i a;
        const int abase = m * LDS_PITCH + kstep * 64 + half * 16;
#pragma unroll
        for (int v = 0; v < 8; ++v) {
            const int koff = (v >> 2) * 32 + (v & 3) * 4;
            a[v] = *(const int*)(&qx[abase + koff]);
        }
        const char* fb0 = fragbase0 + (size_t)kstep * 1024;
        const char* fb1 = fragbase1 + (size_t)kstep * 1024;
        v4i b0lo = *(const v4i*)(fb0);
        v4i b0hi = *(const v4i*)(fb0 + 512);
        v4i b1lo = *(const v4i*)(fb1);
        v4i b1hi = *(const v4i*)(fb1 + 512);
        v8i b0, b1;
#pragma unroll
        for (int v = 0; v < 4; ++v) {
            b0[v] = b0lo[v]; b0[v + 4] = b0hi[v];
            b1[v] = b1lo[v]; b1[v + 4] = b1hi[v];
        }

        acc0 = __builtin_amdgcn_wmma_i32_16x16x64_iu8(true, a, true, b0, acc0, false, false);
        acc1 = __builtin_amdgcn_wmma_i32_16x16x64_iu8(true, a, true, b1, acc1, false, false);
    }

#pragma unroll
    for (int r = 0; r < 8; ++r) {
        const int M = r + half * 8;
        const float sfac = inv_sx[M] * wmean;
        const size_t row = (size_t)(tokbase + M) * OUTN;
        const int n0 = (wave * 2 + 0) * 16 + m;
        const int n1 = n0 + 16;
        out[row + n0] = (float)acc0[r] * sfac + b_lin[n0];
        out[row + n1] = (float)acc1[r] * sfac + b_lin[n1];
    }
}

extern "C" void kernel_launch(void* const* d_in, const int* in_sizes, int n_in,
                              void* d_out, int out_size, void* d_ws, size_t ws_size,
                              hipStream_t stream) {
    (void)in_sizes; (void)n_in; (void)out_size; (void)ws_size;
    const float* x     = (const float*)d_in[0];
    const float* c     = (const float*)d_in[1];
    const float* w_ada = (const float*)d_in[2];
    const float* b_ada = (const float*)d_in[3];
    const float* w_lin = (const float*)d_in[4];
    const float* b_lin = (const float*)d_in[5];
    float* out = (float*)d_out;

    char* ws = (char*)d_ws;
    float*    sum_ws = (float*)ws;
    float*    modws  = (float*)(ws + 256);
    unsigned* wq     = (unsigned*)(ws + 256 + 8 * 2048 * sizeof(float));

    k_wabs_sum<<<1, 256, 0, stream>>>(w_lin, sum_ws);
    k_wquant_swizzle<<<256, 256, 0, stream>>>(w_lin, sum_ws, wq);
    k_adaln_mod<<<2048, 256, 0, stream>>>(c, w_ada, b_ada, modws);
    k_final_bitlinear<<<(BATCH * SEQ) / MTILE, 256, 0, stream>>>(x, modws, wq, sum_ws,
                                                                 b_lin, out);
}
